// TabularConditionedCrossAttention_43258910605536
// MI455X (gfx1250) — matrix operand order repaired and checked by reading, not compiled, not run on hardware
//
#include <hip/hip_runtime.h>

// ---------------------------------------------------------------------------
// TabularConditionedCrossAttention, algebraically simplified.
//
// softmax over a length-1 key axis == 1, so Q/K (and the tabular gating that
// only feeds Q) are dead.  out = concat( ecg @ (Wo_a@Wv_a)^T + (Wo_a@bv_a+bo_a),
//                                        text@ (Wo_b@Wv_b)^T + (Wo_b@bv_b+bo_b) )
// Main work: two 16384x1024x1024 GEMMs -> bf16 WMMA (v_wmma_f32_16x16x32_bf16),
// software-pipelined with double-buffered LDS.
// ---------------------------------------------------------------------------

#define E_DIM 1024
#define B_ROWS 16384

typedef __attribute__((ext_vector_type(4)))  float  v4f;
typedef __attribute__((ext_vector_type(8)))  float  v8f;
typedef __attribute__((ext_vector_type(4)))  __bf16 v4bf;
typedef __attribute__((ext_vector_type(8)))  __bf16 v8bf;
typedef __attribute__((ext_vector_type(16))) __bf16 v16bf;

struct BFPair { v8bf lo, hi; };

__device__ __forceinline__ v16bf make_frag(v8bf lo, v8bf hi) {
  BFPair p{lo, hi};
  return __builtin_bit_cast(v16bf, p);
}

__device__ __forceinline__ v4bf cvt4(v4f v) {
  return v4bf{(__bf16)v.x, (__bf16)v.y, (__bf16)v.z, (__bf16)v.w};
}

// ---------------------------------------------------------------------------
// Transpose + f32->bf16 convert:  dst[n*E + k] = bf16(src[k*E + n])
// ---------------------------------------------------------------------------
__global__ __launch_bounds__(256) void k_transpose_convert(
    const float* __restrict__ src, __bf16* __restrict__ dst, int n) {
  __shared__ __bf16 tile[32][33];
  const int k0 = blockIdx.y * 32, n0 = blockIdx.x * 32;
  const int tx = threadIdx.x & 31, ty = threadIdx.x >> 5;  // 32 x 8
#pragma unroll
  for (int i = 0; i < 32; i += 8)
    tile[ty + i][tx] = (__bf16)src[(size_t)(k0 + ty + i) * n + n0 + tx];
  __syncthreads();
#pragma unroll
  for (int i = 0; i < 32; i += 8)
    dst[(size_t)(n0 + ty + i) * n + k0 + tx] = tile[tx][ty + i];
}

// ---------------------------------------------------------------------------
// bcomb[i] = sum_h Wo[i,h]*bv[h] + bo[i]   (one block per output element)
// ---------------------------------------------------------------------------
__global__ __launch_bounds__(256) void k_bias_compose(
    const float* __restrict__ Wo, const float* __restrict__ bv,
    const float* __restrict__ bo, float* __restrict__ out, int n) {
  __shared__ float red[256];
  const int i = blockIdx.x;
  float s = 0.f;
  for (int h = threadIdx.x; h < n; h += 256) s += Wo[(size_t)i * n + h] * bv[h];
  red[threadIdx.x] = s;
  __syncthreads();
  for (int off = 128; off > 0; off >>= 1) {
    if (threadIdx.x < off) red[threadIdx.x] += red[threadIdx.x + off];
    __syncthreads();
  }
  if (threadIdx.x == 0) out[i] = red[0] + bo[i];
}

// ---------------------------------------------------------------------------
// C[M,N] = A[M,K](f32) * W[N,K](bf16)^T.
// Block tile 128x128 (8 waves as 4Mx2N, each wave 32 rows x 64 cols = 8 accs).
// K pipelined: global loads for tile k+1 issued before WMMAs on tile k;
// double-buffered LDS -> one barrier per K step.
// ---------------------------------------------------------------------------
template <bool BF16_OUT>
__global__ __launch_bounds__(256, 2) void k_gemm_awT(
    const float* __restrict__ A, const __bf16* __restrict__ W,
    const float* __restrict__ bias, float* __restrict__ Cf,
    __bf16* __restrict__ Cb, int M, int N, int K, int ldc, int colOff) {
  constexpr int BM = 128, BN = 128, BK = 32;
  __shared__ alignas(16) __bf16 sA[2][BM * BK];  // 2 x 8 KB
  __shared__ alignas(16) __bf16 sW[2][BN * BK];  // 2 x 8 KB

  const int tid  = threadIdx.x;
  const int wave = tid >> 5;        // 0..7 (wave32)
  const int lane = tid & 31;
  const int mw   = wave >> 1;       // 0..3 : M sub-block of 32 rows
  const int nw   = wave & 1;        // 0..1 : N sub-block of 64 cols
  const int m0 = blockIdx.y * BM;
  const int n0 = blockIdx.x * BN;

  v4f aReg[4];
  v8bf wReg[2];

  auto load_tiles = [&](int k0) {
#pragma unroll
    for (int i = 0; i < 4; ++i) {          // A: 128x32 f32 = 1024 float4
      const int s = tid + 256 * i, row = s >> 3, kc = s & 7;
      aReg[i] = *(const v4f*)(A + (size_t)(m0 + row) * K + k0 + kc * 4);
    }
#pragma unroll
    for (int i = 0; i < 2; ++i) {          // W: 128x32 bf16 = 512 v8bf
      const int s = tid + 256 * i, row = s >> 2, kc = s & 3;
      wReg[i] = *(const v8bf*)(W + (size_t)(n0 + row) * K + k0 + kc * 8);
    }
  };
  auto store_tiles = [&](int buf) {
    __bf16* dA = sA[buf];
    __bf16* dW = sW[buf];
#pragma unroll
    for (int i = 0; i < 4; ++i) {
      const int s = tid + 256 * i, row = s >> 3, kc = s & 7;
      *(v4bf*)(dA + row * BK + kc * 4) = cvt4(aReg[i]);
    }
#pragma unroll
    for (int i = 0; i < 2; ++i) {
      const int s = tid + 256 * i, row = s >> 2, kc = s & 3;
      *(v8bf*)(dW + row * BK + kc * 8) = wReg[i];
    }
  };

  v8f acc[8] = {};  // acc[a*4+j] : a in M (2x16), j in N (4x16)

  load_tiles(0);
  store_tiles(0);
  __syncthreads();

  int cur = 0;
  for (int k0 = 0; k0 < K; k0 += BK) {
    const bool more = (k0 + BK) < K;
    if (more) load_tiles(k0 + BK);   // hide HBM/L2 latency behind WMMAs

    const __bf16* bA = sA[cur];
    const __bf16* bW = sW[cur];
    // A 16x32 frag: lanes<16 hold K0..7 (v0-3) & K16..23 (v4-7); lanes>=16: +8
    v16bf afrag[2];
#pragma unroll
    for (int a = 0; a < 2; ++a) {
      const int arow = mw * 32 + a * 16 + (lane & 15);
      const int akf  = (lane & 16) ? 8 : 0;
      afrag[a] = make_frag(*(const v8bf*)(bA + arow * BK + akf),
                           *(const v8bf*)(bA + arow * BK + akf + 16));
    }
#pragma unroll
    for (int j = 0; j < 4; ++j) {
      // B 32x16 frag: lane%16 = n; lanes<16 hold K0..7 & K16..23, lanes>=16: +8
      const int wrow = nw * 64 + j * 16 + (lane & 15);
      const int wkf  = (lane & 16) ? 8 : 0;
      const v16bf wfrag = make_frag(*(const v8bf*)(bW + wrow * BK + wkf),
                                    *(const v8bf*)(bW + wrow * BK + wkf + 16));
#pragma unroll
      for (int a = 0; a < 2; ++a)
        acc[a * 4 + j] = __builtin_amdgcn_wmma_f32_16x16x32_bf16(
            false, afrag[a], false, wfrag, (short)0, acc[a * 4 + j],
            false, false);
    }

    if (more) store_tiles(cur ^ 1);  // write other buffer; readers are done
    __syncthreads();
    cur ^= 1;
  }

  // C 16x16 f32 layout: VGPR r -> M = r + (lane<16?0:8), N = lane%16
  const int nloc = lane & 15;
  const int mofs = (lane & 16) ? 8 : 0;
#pragma unroll
  for (int a = 0; a < 2; ++a) {
    const int rbase = m0 + mw * 32 + a * 16 + mofs;
#pragma unroll
    for (int j = 0; j < 4; ++j) {
      const int col = n0 + nw * 64 + j * 16 + nloc;
#pragma unroll
      for (int r = 0; r < 8; ++r) {
        const size_t idx = (size_t)(rbase + r) * ldc + colOff + col;
        if constexpr (BF16_OUT) {
          Cb[idx] = (__bf16)acc[a * 4 + j][r];
        } else {
          Cf[idx] = acc[a * 4 + j][r] + bias[col];
        }
      }
    }
  }
}

// ---------------------------------------------------------------------------
extern "C" void kernel_launch(void* const* d_in, const int* in_sizes, int n_in,
                              void* d_out, int out_size, void* d_ws,
                              size_t ws_size, hipStream_t stream) {
  (void)in_sizes; (void)n_in; (void)out_size; (void)ws_size;
  const int E = E_DIM, B = B_ROWS;

  const float* text   = (const float*)d_in[0];
  const float* ecg    = (const float*)d_in[1];
  // d_in[2..6] (tabular features / gating weights) are provably dead.
  const float* Wqkv_a = (const float*)d_in[7];
  const float* bqkv_a = (const float*)d_in[8];
  const float* Wo_a   = (const float*)d_in[9];
  const float* bo_a   = (const float*)d_in[10];
  const float* Wqkv_b = (const float*)d_in[11];
  const float* bqkv_b = (const float*)d_in[12];
  const float* Wo_b   = (const float*)d_in[13];
  const float* bo_b   = (const float*)d_in[14];
  float* out = (float*)d_out;

  char* ws = (char*)d_ws;
  const size_t mat = (size_t)E * E * sizeof(__bf16);
  __bf16* WvT_a   = (__bf16*)ws; ws += mat;
  __bf16* WvT_b   = (__bf16*)ws; ws += mat;
  __bf16* Wcomb_a = (__bf16*)ws; ws += mat;
  __bf16* Wcomb_b = (__bf16*)ws; ws += mat;
  float* bias_a = (float*)ws; ws += E * sizeof(float);
  float* bias_b = (float*)ws; ws += E * sizeof(float);

  const float* Wv_a = Wqkv_a + (size_t)2 * E * E;  // V slice of in_proj
  const float* Wv_b = Wqkv_b + (size_t)2 * E * E;
  const float* bv_a = bqkv_a + 2 * E;
  const float* bv_b = bqkv_b + 2 * E;

  dim3 tgrid(E / 32, E / 32);
  k_transpose_convert<<<tgrid, 256, 0, stream>>>(Wv_a, WvT_a, E);
  k_transpose_convert<<<tgrid, 256, 0, stream>>>(Wv_b, WvT_b, E);

  k_bias_compose<<<E, 256, 0, stream>>>(Wo_a, bv_a, bo_a, bias_a, E);
  k_bias_compose<<<E, 256, 0, stream>>>(Wo_b, bv_b, bo_b, bias_b, E);

  // Wcomb = Wo @ Wv  ==  Wo[M,K] * (Wv^T)[N,K]^T   (bf16 out)
  dim3 cgrid(E / 128, E / 128);
  k_gemm_awT<true><<<cgrid, 256, 0, stream>>>(Wo_a, WvT_a, nullptr, nullptr,
                                              Wcomb_a, E, E, E, E, 0);
  k_gemm_awT<true><<<cgrid, 256, 0, stream>>>(Wo_b, WvT_b, nullptr, nullptr,
                                              Wcomb_b, E, E, E, E, 0);

  // out[:, 0:E)  = ecg  @ Wcomb_a^T + bias_a
  // out[:, E:2E) = text @ Wcomb_b^T + bias_b
  dim3 mgrid(E / 128, B / 128);
  k_gemm_awT<false><<<mgrid, 256, 0, stream>>>(ecg, Wcomb_a, bias_a, out,
                                               nullptr, B, E, E, 2 * E, 0);
  k_gemm_awT<false><<<mgrid, 256, 0, stream>>>(text, Wcomb_b, bias_b, out,
                                               nullptr, B, E, E, 2 * E, E);
}
